// NTM_76613626626119
// MI455X (gfx1250) — matrix operand order repaired and checked by reading, not compiled, not run on hardware
//
#include <hip/hip_runtime.h>

// ---------------------------------------------------------------------------
// NTM forward scan for gfx1250 (MI455X).
// B=16, T=64, IN=64, STATE=1024, H=2, N=1024, M=64, OUT=64, upd=404.
// GEMMs on v_wmma_f32_16x16x32_f16 (f16 in, f32 acc); weights pre-packed once
// into WMMA B-fragment layout. Serial scan: 3 stream-ordered kernels/step
// (state-GEMM, out/upd-GEMM, fused addressing+memwrite+next-comb).
// ---------------------------------------------------------------------------

typedef __attribute__((ext_vector_type(16))) _Float16 v16h;
typedef __attribute__((ext_vector_type(8)))  _Float16 v8h;
typedef __attribute__((ext_vector_type(8)))  float    v8f;

#define NTM_B 16
#define NTM_T 64
#define NTM_IN 64
#define NTM_STATE 1024
#define NTM_H 2
#define NTM_N 1024
#define NTM_M 64
#define NTM_OUT 64
#define NTM_FAN1 1216          // IN + STATE + H*M
#define NTM_UPD 404            // H * 202
#define NTM_UPD_PAD 416        // 26 tiles of 16
#define NTM_EPS 1e-8f

__device__ __forceinline__ float ntm_sigmoid(float x) { return 1.f / (1.f + expf(-x)); }
__device__ __forceinline__ float ntm_softplus(float x) { return (x > 20.f) ? x : log1pf(expf(x)); }

// -------------------- weight pre-pack: f32 row-major -> f16 WMMA B-fragments
// lane<16 : column N=lane,    K = kstep*32 + 0..15 (16 consecutive halfs)
// lane>=16: column N=lane-16, K = kstep*32 + 16..31
__global__ void ntm_pack_w_f16(const float* __restrict__ W, _Float16* __restrict__ out,
                               int K, int Ncols, int ntiles, int ksteps) {
  int idx = blockIdx.x * blockDim.x + threadIdx.x;
  int total = ntiles * ksteps * 32;
  if (idx >= total) return;
  int lane  = idx & 31;
  int rest  = idx >> 5;
  int kstep = rest % ksteps;
  int tile  = rest / ksteps;
  int col   = tile * 16 + (lane & 15);
  int kbase = kstep * 32 + ((lane & 16) ? 16 : 0);
  _Float16* dst = out + (size_t)idx * 16;
#pragma unroll
  for (int i = 0; i < 16; ++i) {
    int kk = kbase + i;
    float v = (col < Ncols && kk < K) ? W[(size_t)kk * Ncols + col] : 0.f;
    dst[i] = (_Float16)v;
  }
}

// -------------------- per-wave GEMM helper: C(16x16) += A(16xK) * B(Kx16)
// A row-major f16. A-fragment (contiguous k order, same as the B pack):
// lane<16 : row=lane,    halfs 0..15=K{ks*32+0..15}
// lane>=16: row=lane-16, halfs 0..15=K{ks*32+16..31}
__device__ __forceinline__ v8f ntm_wmma_rowtile(const _Float16* __restrict__ A, int lda,
                                                const _Float16* __restrict__ Bp,
                                                int ksteps, int lane) {
  v8f c = {};
  int row   = lane & 15;
  int abase = (lane & 16) ? 16 : 0;
  const _Float16* arow = A + (size_t)row * lda + abase;
  union AFrag { v16h v; v8h h[2]; };
  for (int ks = 0; ks < ksteps; ++ks) {
    AFrag a;
    const _Float16* ap = arow + ks * 32;
    a.h[0] = *(const v8h*)(ap);
    a.h[1] = *(const v8h*)(ap + 8);
    v16h b = *(const v16h*)(Bp + ((size_t)ks * 32 + lane) * 16);
    c = __builtin_amdgcn_wmma_f32_16x16x32_f16(false, a.v, false, b,
                                               (short)0, c, false, false);
  }
  return c;
}

// -------------------- prologue: build comb for t=0 (read via gather-dot)
__global__ void ntm_step_pre(const float* __restrict__ x, const float* __restrict__ ctrl32,
                             const float* __restrict__ wt, const float* __restrict__ mem,
                             _Float16* __restrict__ comb16, int t) {
  int idx = blockIdx.x * blockDim.x + threadIdx.x;
  if (idx >= NTM_B * NTM_FAN1) return;
  int b = idx / NTM_FAN1, col = idx % NTM_FAN1;
  float v;
  if (col < NTM_IN) {
    v = x[((size_t)b * NTM_T + t) * NTM_IN + col];
  } else if (col < NTM_IN + NTM_STATE) {
    v = ctrl32[b * NTM_STATE + (col - NTM_IN)];
  } else {
    int c2 = col - (NTM_IN + NTM_STATE);
    int h = c2 >> 6, m = c2 & 63;
    const float* wrow = wt + ((size_t)b * NTM_H + h) * NTM_N;
    const float* mb   = mem + (size_t)b * NTM_N * NTM_M + m;
    __builtin_prefetch(wrow, 0, 1);   // global_prefetch_b8
    float acc = 0.f;
    for (int n = 0; n < NTM_N; ++n) acc += wrow[n] * mb[(size_t)n * NTM_M];
    v = acc;
  }
  comb16[idx] = (_Float16)v;
}

// -------------------- ctrl = sigmoid(comb @ W_state + b); 64 waves
__global__ void ntm_gemm_state(const _Float16* __restrict__ comb16,
                               const _Float16* __restrict__ Wp,
                               const float* __restrict__ b_state,
                               float* __restrict__ ctrl32, _Float16* __restrict__ ctrl16) {
  int wave = (blockIdx.x * blockDim.x + threadIdx.x) >> 5;
  int lane = threadIdx.x & 31;
  if (wave >= 64) return;  // wave-uniform
  const int ksteps = NTM_FAN1 / 32;  // 38
  v8f c = ntm_wmma_rowtile(comb16, NTM_FAN1, Wp + (size_t)wave * ksteps * 32 * 16,
                           ksteps, lane);
  int col  = wave * 16 + (lane & 15);
  float bias = b_state[col];
  int rbase = (lane & 16) ? 8 : 0;
#pragma unroll
  for (int r = 0; r < 8; ++r) {
    int row = rbase + r;
    float s = ntm_sigmoid(c[r] + bias);
    ctrl32[row * NTM_STATE + col] = s;
    ctrl16[row * NTM_STATE + col] = (_Float16)s;
  }
}

// -------------------- out = ctrl@W_out+b ; upd = ctrl@W_upd+b; 30 waves
__global__ void ntm_gemm_outupd(const _Float16* __restrict__ ctrl16,
                                const _Float16* __restrict__ WpO,
                                const _Float16* __restrict__ WpU,
                                const float* __restrict__ b_out,
                                const float* __restrict__ b_upd,
                                float* __restrict__ y, int t,
                                float* __restrict__ upd32) {
  int wave = (blockIdx.x * blockDim.x + threadIdx.x) >> 5;
  int lane = threadIdx.x & 31;
  if (wave >= 30) return;  // wave-uniform
  const int ksteps = NTM_STATE / 32;  // 32
  const size_t tilesz = (size_t)ksteps * 32 * 16;
  const _Float16* Bp = (wave < 4) ? (WpO + (size_t)wave * tilesz)
                                  : (WpU + (size_t)(wave - 4) * tilesz);
  v8f c = ntm_wmma_rowtile(ctrl16, NTM_STATE, Bp, ksteps, lane);
  int rbase = (lane & 16) ? 8 : 0;
  if (wave < 4) {
    int col = wave * 16 + (lane & 15);
    float bias = b_out[col];
#pragma unroll
    for (int r = 0; r < 8; ++r) {
      int row = rbase + r;
      y[((size_t)row * NTM_T + t) * NTM_OUT + col] = c[r] + bias;
    }
  } else {
    int col = (wave - 4) * 16 + (lane & 15);
    if (col < NTM_UPD) {
      float bias = b_upd[col];
#pragma unroll
      for (int r = 0; r < 8; ++r)
        upd32[(rbase + r) * NTM_UPD_PAD + col] = c[r] + bias;
    }
  }
}

// -------------------- segmented block reductions (256-thread halves)
__device__ __forceinline__ float ntm_red_max2(float v, float* sh, int lt) {
  sh[lt] = v; __syncthreads();
  for (int s = 128; s >= 1; s >>= 1) { if (lt < s) sh[lt] = fmaxf(sh[lt], sh[lt + s]); __syncthreads(); }
  float r = sh[0]; __syncthreads(); return r;
}
__device__ __forceinline__ float ntm_red_sum2(float v, float* sh, int lt) {
  sh[lt] = v; __syncthreads();
  for (int s = 128; s >= 1; s >>= 1) { if (lt < s) sh[lt] += sh[lt + s]; __syncthreads(); }
  float r = sh[0]; __syncthreads(); return r;
}

// -------------------- fused: addressing (both heads) + erase/add memory write
// + next-step comb assembly. One block per batch element, 512 threads.
// upd per head: k[64] beta g s[3] gamma erase[64] add[64] jd j[3]
__global__ __launch_bounds__(512)
void ntm_fused(const float* __restrict__ upd32, float* __restrict__ mem,
               float* __restrict__ wt, float* __restrict__ wtd,
               const float* __restrict__ ctrl32, const float* __restrict__ x,
               _Float16* __restrict__ comb16, int t, int build_next) {
  __shared__ float s_wg[NTM_H][NTM_N];      // wg, later reused for wt_fin
  __shared__ float s_wp[NTM_H][NTM_N];
  __shared__ float s_red[NTM_H][256];
  __shared__ float s_sc[NTM_H][12];
  __shared__ float s_esg[NTM_H][NTM_M];
  __shared__ float s_eadd[NTM_H][NTM_M];
  __shared__ float s_part[NTM_H][8][NTM_M]; // read-einsum partials

  int b   = blockIdx.x;
  int tid = threadIdx.x;
  int h   = tid >> 8;        // head = thread half
  int lt  = tid & 255;       // lane within head
  const float* u = upd32 + (size_t)b * NTM_UPD_PAD + h * 202;
  float* memb = mem + (size_t)b * NTM_N * NTM_M;
  __builtin_prefetch(memb, 0, 1);   // global_prefetch_b8

  if (lt == 0) {
    s_sc[h][0] = ntm_softplus(u[64]);          // beta
    s_sc[h][1] = ntm_sigmoid(u[65]);           // g
    float a0 = u[66], a1 = u[67], a2 = u[68];
    float mx = fmaxf(a0, fmaxf(a1, a2));
    float e0 = expf(a0 - mx), e1 = expf(a1 - mx), e2 = expf(a2 - mx);
    float inv = 1.f / (e0 + e1 + e2);
    s_sc[h][2] = e0 * inv; s_sc[h][3] = e1 * inv; s_sc[h][4] = e2 * inv; // s
    s_sc[h][5] = 1.f + ntm_softplus(u[69]);    // gamma
    s_sc[h][6] = ntm_sigmoid(u[198]);          // jd
    float c0 = u[199], c1 = u[200], c2 = u[201];
    float mj = fmaxf(c0, fmaxf(c1, c2));
    float f0 = expf(c0 - mj), f1 = expf(c1 - mj), f2 = expf(c2 - mj);
    float invj = 1.f / (f0 + f1 + f2);
    s_sc[h][7] = f0 * invj; s_sc[h][8] = f1 * invj; s_sc[h][9] = f2 * invj; // j
    float kn = 0.f;
    for (int i = 0; i < NTM_M; ++i) { float kv = u[i]; kn += kv * kv; }
    s_sc[h][10] = sqrtf(kn);                   // ||k||
  }
  if (lt < NTM_M) {
    s_esg[h][lt]  = ntm_sigmoid(u[70 + lt]);
    s_eadd[h][lt] = u[134 + lt];
  }
  __syncthreads();
  float beta = s_sc[h][0], g = s_sc[h][1];
  float s0 = s_sc[h][2], s1 = s_sc[h][3], s2 = s_sc[h][4];
  float gamma = s_sc[h][5], jd = s_sc[h][6];
  float j0 = s_sc[h][7], j1 = s_sc[h][8], j2 = s_sc[h][9];
  float knorm = s_sc[h][10];

  float* wtbh  = wt  + ((size_t)b * NTM_H + h) * NTM_N;
  float* wtdbh = wtd + ((size_t)b * NTM_H + h) * NTM_N;

  // cosine-similarity logits (4 n's per thread per head)
  float z[4];
  float lmax = -1e30f;
#pragma unroll
  for (int q = 0; q < 4; ++q) {
    int n = lt + q * 256;
    const float* mrow = memb + (size_t)n * NTM_M;
    float dot = 0.f, nn = 0.f;
    for (int i = 0; i < NTM_M; ++i) { float mv = mrow[i]; dot += u[i] * mv; nn += mv * mv; }
    z[q] = beta * (dot / (knorm * sqrtf(nn) + NTM_EPS));
    lmax = fmaxf(lmax, z[q]);
  }
  float mx = ntm_red_max2(lmax, s_red[h], lt);
  float lsum = 0.f;
#pragma unroll
  for (int q = 0; q < 4; ++q) { z[q] = expf(z[q] - mx); lsum += z[q]; }
  float inv = 1.f / ntm_red_sum2(lsum, s_red[h], lt);
  // gated interpolation with previous attention
#pragma unroll
  for (int q = 0; q < 4; ++q) {
    int n = lt + q * 256;
    s_wg[h][n] = g * (z[q] * inv) + (1.f - g) * wtbh[n];
  }
  __syncthreads();
  // circular shift + sharpen
  float lsum2 = 0.f;
#pragma unroll
  for (int q = 0; q < 4; ++q) {
    int n = lt + q * 256;
    float wsv = s0 * s_wg[h][(n + 1) & (NTM_N - 1)] + s1 * s_wg[h][n]
              + s2 * s_wg[h][(n + NTM_N - 1) & (NTM_N - 1)];
    float wpv = expf(gamma * logf(wsv + NTM_EPS));
    s_wp[h][n] = wpv;
    lsum2 += wpv;
  }
  float inv2 = 1.f / ntm_red_sum2(lsum2, s_red[h], lt);
  // bookmark update + 3-way jump; stash wt_fin in s_wg for the mem phase
#pragma unroll
  for (int q = 0; q < 4; ++q) {
    int n = lt + q * 256;
    float wtn  = s_wp[h][n] * inv2;
    float wtdn = (1.f - jd) * wtdbh[n] + jd * wtn;
    float w0   = (n == 0) ? 1.f : 0.f;
    float wtf  = j0 * wtn + j1 * wtdn + j2 * w0;
    wtdbh[n] = wtdn;
    wtbh[n]  = wtf;
    s_wg[h][n] = wtf;
  }
  __syncthreads();

  // erase/add memory write, fused with next-step read accumulation.
  // 512 | 64*1024 => each thread owns fixed column m = tid&63.
  int m = tid & 63;
  float eg0 = s_esg[0][m], eg1 = s_esg[1][m];
  float ad0 = s_eadd[0][m], ad1 = s_eadd[1][m];
  float racc0 = 0.f, racc1 = 0.f;
#pragma unroll 4
  for (int k = 0; k < (NTM_N * NTM_M) / 512; ++k) {
    int idx = tid + 512 * k;
    int n = idx >> 6;
    float w0 = s_wg[0][n], w1 = s_wg[1][n];
    float nv = memb[idx] * ((1.f - w0 * eg0) * (1.f - w1 * eg1))
             + (w0 * ad0 + w1 * ad1);
    memb[idx] = nv;
    racc0 += w0 * nv;
    racc1 += w1 * nv;
  }
  s_part[0][tid >> 6][m] = racc0;
  s_part[1][tid >> 6][m] = racc1;
  __syncthreads();

  if (build_next) {
    // read columns of next comb: reduce the 8 partials per (h,m)
    if (tid < NTM_H * NTM_M) {
      int hh = tid >> 6, mm = tid & 63;
      float r = 0.f;
#pragma unroll
      for (int j = 0; j < 8; ++j) r += s_part[hh][j][mm];
      comb16[(size_t)b * NTM_FAN1 + NTM_IN + NTM_STATE + hh * NTM_M + mm] = (_Float16)r;
    }
    // x_{t+1} and ctrl columns
    for (int col = tid; col < NTM_IN + NTM_STATE; col += 512) {
      float v = (col < NTM_IN)
              ? x[((size_t)b * NTM_T + (t + 1)) * NTM_IN + col]
              : ctrl32[b * NTM_STATE + (col - NTM_IN)];
      comb16[(size_t)b * NTM_FAN1 + col] = (_Float16)v;
    }
  }
}

// ---------------------------------------------------------------------------
extern "C" void kernel_launch(void* const* d_in, const int* in_sizes, int n_in,
                              void* d_out, int out_size, void* d_ws, size_t ws_size,
                              hipStream_t stream) {
  (void)in_sizes; (void)n_in; (void)out_size; (void)ws_size;
  const float* x       = (const float*)d_in[0];
  const float* ctrl0   = (const float*)d_in[1];
  const float* wt0     = (const float*)d_in[2];
  const float* wtd0    = (const float*)d_in[3];
  const float* mem0    = (const float*)d_in[4];
  const float* W_state = (const float*)d_in[5];
  const float* b_state = (const float*)d_in[6];
  const float* W_out   = (const float*)d_in[7];
  const float* b_out   = (const float*)d_in[8];
  const float* W_upd   = (const float*)d_in[9];
  const float* b_upd   = (const float*)d_in[10];

  // d_out = [output(16*64*64) | ctrl_f(16*1024) | wt_f(16*2*1024) | mem_f(16*1024*64)]
  float* y    = (float*)d_out;
  float* ctrl = y + NTM_B * NTM_T * NTM_OUT;   // live ctrl state
  float* wt   = ctrl + NTM_B * NTM_STATE;      // live attention
  float* mem  = wt + NTM_B * NTM_H * NTM_N;    // live memory

  // workspace carve-out (256B aligned)
  char* ws = (char*)d_ws;
  size_t off = 0;
  auto carve = [&](size_t bytes) { void* p = ws + off; off = (off + bytes + 255) & ~(size_t)255; return p; };
  _Float16* packWs = (_Float16*)carve((size_t)64 * 38 * 32 * 16 * 2);
  _Float16* packWu = (_Float16*)carve((size_t)26 * 32 * 32 * 16 * 2);
  _Float16* packWo = (_Float16*)carve((size_t)4  * 32 * 32 * 16 * 2);
  _Float16* comb16 = (_Float16*)carve((size_t)NTM_B * NTM_FAN1 * 2);
  _Float16* ctrl16 = (_Float16*)carve((size_t)NTM_B * NTM_STATE * 2);
  float* upd32 = (float*)carve((size_t)NTM_B * NTM_UPD_PAD * 4);
  float* wtd   = (float*)carve((size_t)NTM_B * NTM_H * NTM_N * 4);

  // initialize live state from inputs (deterministic per launch)
  hipMemcpyAsync(ctrl, ctrl0, (size_t)NTM_B * NTM_STATE * 4, hipMemcpyDeviceToDevice, stream);
  hipMemcpyAsync(wt,   wt0,   (size_t)NTM_B * NTM_H * NTM_N * 4, hipMemcpyDeviceToDevice, stream);
  hipMemcpyAsync(wtd,  wtd0,  (size_t)NTM_B * NTM_H * NTM_N * 4, hipMemcpyDeviceToDevice, stream);
  hipMemcpyAsync(mem,  mem0,  (size_t)NTM_B * NTM_N * NTM_M * 4, hipMemcpyDeviceToDevice, stream);

  // pack weights once per launch
  { int tot = 64 * 38 * 32; ntm_pack_w_f16<<<(tot + 255) / 256, 256, 0, stream>>>(W_state, packWs, NTM_FAN1, NTM_STATE, 64, 38); }
  { int tot = 26 * 32 * 32; ntm_pack_w_f16<<<(tot + 255) / 256, 256, 0, stream>>>(W_upd, packWu, NTM_STATE, NTM_UPD, 26, 32); }
  { int tot = 4  * 32 * 32; ntm_pack_w_f16<<<(tot + 255) / 256, 256, 0, stream>>>(W_out, packWo, NTM_STATE, NTM_OUT, 4, 32); }

  // prologue: comb for t=0
  ntm_step_pre<<<(NTM_B * NTM_FAN1 + 255) / 256, 256, 0, stream>>>(x, ctrl, wt, mem, comb16, 0);

  // sequential scan over T: 3 kernels per step
  for (int t = 0; t < NTM_T; ++t) {
    ntm_gemm_state<<<16, 128, 0, stream>>>(comb16, packWs, b_state, ctrl, ctrl16);
    ntm_gemm_outupd<<<8, 128, 0, stream>>>(ctrl16, packWo, packWu, b_out, b_upd, y, t, upd32);
    ntm_fused<<<NTM_B, 512, 0, stream>>>(upd32, mem, wt, wtd, ctrl, x, comb16, t,
                                         (t + 1 < NTM_T) ? 1 : 0);
  }
}
